// BN_ReLU_DConvCeption_27049704030778
// MI455X (gfx1250) — matrix operand order repaired and checked by reading, not compiled, not run on hardware
//
#include <hip/hip_runtime.h>

// Problem constants
#define BB   8
#define CIN  128
#define HH   64
#define WW   64
#define HW   4096            // HH*WW
#define KKR  1152            // CIN*9 reduction length
#define KST  36              // KKR/32 WMMA K-steps
#define NPIX 16              // pixels per M-tile
#define LROW 1160            // 1152 + 8 halfword pad (2320B row -> conflict-free b128)

typedef __attribute__((ext_vector_type(16))) _Float16 v16h;
typedef __attribute__((ext_vector_type(8)))  float    v8f;

union AFrag { v16h v; uint4 q[2]; };

// ---------------- BN + ReLU (float4 vectorized; HW=4096 so no channel straddle) ----------------
__global__ void bn_relu_kernel(const float4* __restrict__ x,
                               const float* __restrict__ gamma,
                               const float* __restrict__ beta,
                               const float* __restrict__ mean,
                               const float* __restrict__ var,
                               float4* __restrict__ h) {
  int i = blockIdx.x * blockDim.x + threadIdx.x;
  if (i >= (BB * CIN * HW) / 4) return;
  int c = (i >> 10) & (CIN - 1);            // 1024 float4 per channel plane
  float inv = rsqrtf(var[c] + 1e-5f);
  float s = inv * gamma[c];
  float m = mean[c];
  float be = beta[c];
  float4 v = x[i];
  float4 r;
  r.x = fmaxf((v.x - m) * s + be, 0.f);
  r.y = fmaxf((v.y - m) * s + be, 0.f);
  r.z = fmaxf((v.z - m) * s + be, 0.f);
  r.w = fmaxf((v.w - m) * s + be, 0.f);
  h[i] = r;
}

// ------------- f32 weights -> f16, padded to [co_pad][K=c*9+k2] -------------
__global__ void prep_w_kernel(const float* __restrict__ w, _Float16* __restrict__ Wt,
                              int co_act, int co_pad) {
  int i = blockIdx.x * blockDim.x + threadIdx.x;
  if (i >= KKR * co_pad) return;
  int k  = i / co_pad;
  int co = i - k * co_pad;
  float v = 0.f;
  if (co < co_act) v = w[(size_t)co * KKR + k];   // w[co][c][ky][kx] is flat co*1152 + k
  Wt[(size_t)co * KKR + k] = (_Float16)v;
}

// ------------- unified deformable-conv implicit GEMM (offset==null => plain conv) -------------
__global__ __launch_bounds__(256)
void deform_gemm_kernel(const float* __restrict__ h,
                        const float* __restrict__ off,      // [B][18][H][W] or nullptr
                        const _Float16* __restrict__ Wt,    // [co_pad][KKR] f16
                        const float* __restrict__ bias,     // [co_act]
                        float* __restrict__ out,            // [B][co_act][H][W]
                        int co_act, int co_pad) {
  __shared__ __align__(16) _Float16 A[NPIX * LROW];
  __shared__ float OFF[NPIX * 18];

  const int tid = threadIdx.x;
  const int nb  = blockIdx.x * NPIX;        // 16 consecutive pixels within one row
  const int b   = nb >> 12;
  const int rem = nb & 4095;
  const int y   = rem >> 6;
  const int x0  = rem & 63;

  // hint: prefetch Wt + 16*wave halfwords (inside co row 0) toward the caches
  __builtin_prefetch(Wt + ((tid >> 5) << 4), 0, 1);

  // stage 18 offset channels for the 16 pixels
  if (tid < NPIX * 18) {
    int p = tid / 18, r = tid - p * 18;     // r = k2*2 + {dy,dx}
    float v = 0.f;
    if (off) v = off[(((size_t)b * 18 + r) * HH + y) * WW + x0 + p];
    OFF[p * 18 + r] = v;
  }
  __syncthreads();

  // ---- bilinear sampling into LDS A-tile (f16), lane->pixel so gathers coalesce ----
  // Corner indices are clamped in-bounds; invalid/zero corners are annihilated by
  // zero weights, so loads are UNCONDITIONAL (straight-line, clause-able) — no exec
  // mask churn per corner.
  {
    const int p  = tid & 15;                // pixel
    const int kc = tid >> 4;                // channel chunk: 8 channels each
    const int xp = x0 + p;
    const float* hb = h + (size_t)b * (CIN * HW) + (size_t)(kc * 8) * HW;
    #pragma unroll
    for (int k2 = 0; k2 < 9; ++k2) {
      const int ky = k2 / 3, kx = k2 - ky * 3;
      const float dy = OFF[p * 18 + k2 * 2 + 0];
      const float dx = OFF[p * 18 + k2 * 2 + 1];
      const float py = (float)(y - 1 + ky) + dy;
      const float px = (float)(xp - 1 + kx) + dx;
      const float fy = floorf(py), fx = floorf(px);
      const int iy0 = (int)fy, ix0 = (int)fx;
      const float wy1 = py - fy, wx1 = px - fx;
      const float wy0 = 1.f - wy1, wx0 = 1.f - wx1;
      const bool vy0 = (iy0 >= 0) & (iy0 < HH);
      const bool vy1 = (iy0 + 1 >= 0) & (iy0 + 1 < HH);
      const bool vx0 = (ix0 >= 0) & (ix0 < WW);
      const bool vx1 = (ix0 + 1 >= 0) & (ix0 + 1 < WW);
      const int yc0 = min(max(iy0, 0), HH - 1), yc1 = min(max(iy0 + 1, 0), HH - 1);
      const int xc0 = min(max(ix0, 0), WW - 1), xc1 = min(max(ix0 + 1, 0), WW - 1);
      const float w00 = (vy0 & vx0) ? wy0 * wx0 : 0.f;
      const float w01 = (vy0 & vx1) ? wy0 * wx1 : 0.f;
      const float w10 = (vy1 & vx0) ? wy1 * wx0 : 0.f;
      const float w11 = (vy1 & vx1) ? wy1 * wx1 : 0.f;
      const int i00 = yc0 * WW + xc0, i01 = yc0 * WW + xc1;
      const int i10 = yc1 * WW + xc0, i11 = yc1 * WW + xc1;
      _Float16* arow = &A[p * LROW + (kc * 8) * 9 + k2];
      const float* hc = hb;
      #pragma unroll
      for (int cc = 0; cc < 8; ++cc) {
        float v;
        v = w00 * hc[i00];
        v = fmaf(w01, hc[i01], v);
        v = fmaf(w10, hc[i10], v);
        v = fmaf(w11, hc[i11], v);
        arow[cc * 9] = (_Float16)v;
        hc += HW;
      }
    }
  }
  __syncthreads();

  // ---- per-wave 16x16 GEMM tile over K=1152 in 36 WMMA steps ----
  const int wv   = tid >> 5;
  const int lane = tid & 31;
  const int cobase = wv * 16;
  if (cobase >= co_pad) return;
  const int lh = lane >> 4;                 // wave half
  const int lm = lane & 15;                 // M row (A) / N col (D)

  const uint4* Aq = reinterpret_cast<const uint4*>(A);
  const int abase = lm * (LROW / 8) + lh;   // LROW/8 = 145 uint4 per row
  const uint4* Bq = reinterpret_cast<const uint4*>(Wt + (size_t)(cobase + lm) * KKR) + lh;  // B column n per lane

  v8f acc = {0.f, 0.f, 0.f, 0.f, 0.f, 0.f, 0.f, 0.f};
  #pragma unroll 2
  for (int kk = 0; kk < KST; ++kk) {
    AFrag af;
    // A 16x32 f16 layout: lanes0-15 K {0..7,16..23}, lanes16-31 K {8..15,24..31}
    af.q[0] = Aq[abase + kk * 4];
    af.q[1] = Aq[abase + kk * 4 + 2];
    // B 32x16 f16: lane holds N col, K in same order as A
    AFrag bf;
    bf.q[0] = Bq[kk * 4];
    bf.q[1] = Bq[kk * 4 + 2];
    acc = __builtin_amdgcn_wmma_f32_16x16x32_f16(false, af.v, false, bf.v,
                                                 (short)0, acc, false, false);
  }

  // ---- epilogue: D layout lane=N(co), VGPR v = M (pixel), +8 for upper half ----
  const int co = cobase + lm;
  if (co < co_act) {
    const float bs = bias[co];
    float* orow = out + (((size_t)b * co_act + co) * HH + y) * WW + x0;
    #pragma unroll
    for (int v = 0; v < 8; ++v) {
      orow[v + 8 * lh] = acc[v] + bs;
    }
  }
}

extern "C" void kernel_launch(void* const* d_in, const int* in_sizes, int n_in,
                              void* d_out, int out_size, void* d_ws, size_t ws_size,
                              hipStream_t stream) {
  const float* x    = (const float*)d_in[0];
  const float* g    = (const float*)d_in[1];
  const float* be   = (const float*)d_in[2];
  const float* mu   = (const float*)d_in[3];
  const float* var  = (const float*)d_in[4];
  const float* offw = (const float*)d_in[5];
  const float* offb = (const float*)d_in[6];
  const float* odcw = (const float*)d_in[7];
  const float* odcb = (const float*)d_in[8];
  const float* dcw  = (const float*)d_in[9];
  const float* dcb  = (const float*)d_in[10];
  (void)in_sizes; (void)n_in; (void)out_size; (void)ws_size;

  char* ws = (char*)d_ws;
  float*    h     = (float*)(ws);                          // 16 MB
  float*    coff  = (float*)(ws + 16777216);               // [8][18][64][64]
  float*    doff  = (float*)(ws + 16777216 + 2359296);     // [8][18][64][64]
  _Float16* WtOff = (_Float16*)(ws + 21495808);            // [32][1152]
  _Float16* WtOdc = (_Float16*)(ws + 21495808 + 73728);
  _Float16* WtDc  = (_Float16*)(ws + 21495808 + 147456);   // [128][1152]

  bn_relu_kernel<<<4096, 256, 0, stream>>>((const float4*)x, g, be, mu, var, (float4*)h);
  prep_w_kernel<<<144, 256, 0, stream>>>(offw, WtOff, 18, 32);
  prep_w_kernel<<<144, 256, 0, stream>>>(odcw, WtOdc, 18, 32);
  prep_w_kernel<<<576, 256, 0, stream>>>(dcw,  WtDc, 128, 128);

  // pass 1: offset==null => standard 3x3 conv
  deform_gemm_kernel<<<2048, 256, 0, stream>>>(h, nullptr, WtOff, offb, coff, 18, 32);
  // pass 2: deform conv with conv_offsets -> dconv_offsets
  deform_gemm_kernel<<<2048, 256, 0, stream>>>(h, coff, WtOdc, odcb, doff, 18, 32);
  // pass 3: deform conv with dconv_offsets -> final output
  deform_gemm_kernel<<<2048, 256, 0, stream>>>(h, doff, WtDc, dcb, (float*)d_out, 128, 128);
}
